// ELPHEdgeAware_44160853737921
// MI455X (gfx1250) — compile-verified, matrix operand order checked by reading, not run on hardware
//
#include <hip/hip_runtime.h>
#include <math.h>

#define NN 100000
#define NE 800000
#define DN 64
#define HM 128
#define HU 128

typedef __attribute__((ext_vector_type(16))) _Float16 v16h;
typedef __attribute__((ext_vector_type(8)))  _Float16 v8h;
typedef __attribute__((ext_vector_type(4)))  _Float16 v4h;
typedef __attribute__((ext_vector_type(8)))  float    v8f;

// Workspace layout (in halves). Fragment = 512 halves: 32 lanes x 16 contiguous halves.
#define W1F_OFF 0       // 5 kchunks * 8 ntiles * 512 = 20480
#define W2F_OFF 20480   // 4 * 4 * 512 = 8192
#define U1F_OFF 28672   // 4 * 8 * 512 = 16384
#define U2F_OFF 45056   // 4 * 4 * 512 = 8192
#define WSF_TOTAL 53248

__device__ __forceinline__ v8f wmma_f16(v16h a, v16h b, v8f c) {
  // (neg_a, A, neg_b, B, c_mod, C, reuse_a, reuse_b)
  return __builtin_amdgcn_wmma_f32_16x16x32_f16(false, a, false, b, (short)0, c,
                                                false, false);
}

// ---------------- weight pre-swizzle (f32 row-major -> f16 fragment layout) ---------
__device__ __forceinline__ void prep_frag_elem(const float* __restrict__ W,
                                               _Float16* __restrict__ dst,
                                               int fid, int NT, int Ncols, int Kact) {
  int frag   = fid >> 9;
  int within = fid & 511;
  int l = within >> 4;
  int i = within & 15;
  int c = frag / NT;
  int t = frag - c * NT;
  int col = l & 15, khi = l >> 4;
  int v = i >> 1, hh = i & 1;
  // CDNA5 16-bit operand layout: half i of lane l holds K below (within 32-K chunk c)
  int k = 32 * c + ((v < 4) ? (2 * v + hh + 8 * khi)
                            : (16 + 2 * (v - 4) + hh + 8 * khi));
  int n = 16 * t + col;
  float val = (k < Kact) ? W[(size_t)k * Ncols + n] : 0.0f;
  dst[fid] = (_Float16)val;
}

__global__ void prep_weights(const float* __restrict__ W1, const float* __restrict__ W2,
                             const float* __restrict__ U1, const float* __restrict__ U2,
                             _Float16* __restrict__ wsf) {
  int tid = blockIdx.x * blockDim.x + threadIdx.x;
  if (tid < W2F_OFF) {
    prep_frag_elem(W1, wsf + W1F_OFF, tid - W1F_OFF, 8, HM, 132);   // [132->160] x 128
  } else if (tid < U1F_OFF) {
    prep_frag_elem(W2, wsf + W2F_OFF, tid - W2F_OFF, 4, DN, 128);   // 128 x 64
  } else if (tid < U2F_OFF) {
    prep_frag_elem(U1, wsf + U1F_OFF, tid - U1F_OFF, 8, HU, 128);   // 128 x 128
  } else if (tid < WSF_TOTAL) {
    prep_frag_elem(U2, wsf + U2F_OFF, tid - U2F_OFF, 4, DN, 128);   // 128 x 64
  }
}

// ---------------- edge message + scatter-add -----------------------------------------
#define EWAVES 4
__global__ __launch_bounds__(EWAVES * 32) void edge_msg_kernel(
    const float* __restrict__ x, const int* __restrict__ eidx,
    const float* __restrict__ ef, const float* __restrict__ b1,
    const float* __restrict__ b2, const _Float16* __restrict__ wsf,
    float* agg) {
  __shared__ __align__(16) _Float16 aStage[EWAVES][16][160];
  __shared__ __align__(16) float    hStage[EWAVES][16][128];

  const int lane = threadIdx.x & 31;
  const int w    = threadIdx.x >> 5;
  const int col  = lane & 15;
  const int khi  = lane >> 4;
  const int gwave  = blockIdx.x * EWAVES + w;
  const int nwaves = gridDim.x * EWAVES;

  const _Float16* w1f = wsf + W1F_OFF;
  const _Float16* w2f = wsf + W2F_OFF;

  float b1v[8], b2v[4];
  for (int t = 0; t < 8; ++t) b1v[t] = b1[t * 16 + col];
  for (int t = 0; t < 4; ++t) b2v[t] = b2[t * 16 + col];

  // zero the K-pad region [132,160) once (never overwritten)
  if (lane < 16) {
    v4h z = {(_Float16)0.f, (_Float16)0.f, (_Float16)0.f, (_Float16)0.f};
    for (int j = 0; j < 7; ++j) *(v4h*)&aStage[w][lane][132 + 4 * j] = z;
  }

  const int NTILES = NE / 16;  // 50000, exact
  for (int tile = gwave; tile < NTILES; tile += nwaves) {
    const int base = tile * 16;

    // ---- gather: lane covers row m=col, half khi (0: x[src], 1: x[dst]) ----
    {
      const int m = col;
      const int e = base + m;
      const unsigned idx = (unsigned)eidx[khi * NE + e];
      const float* xr = x + (idx << 6);
      _Float16* drow = &aStage[w][m][khi * 64];
      for (int j = 0; j < 8; ++j) {
        float4 f0 = *(const float4*)(xr + 8 * j);
        float4 f1 = *(const float4*)(xr + 8 * j + 4);
        v8h hv = {(_Float16)f0.x, (_Float16)f0.y, (_Float16)f0.z, (_Float16)f0.w,
                  (_Float16)f1.x, (_Float16)f1.y, (_Float16)f1.z, (_Float16)f1.w};
        *(v8h*)(drow + 8 * j) = hv;
      }
      if (khi == 0) {
        float4 e4 = *(const float4*)(ef + (unsigned)e * 4u);
        v4h ev = {(_Float16)log1pf(e4.x), (_Float16)log1pf(e4.y),
                  (_Float16)log1pf(e4.z), (_Float16)log1pf(e4.w)};
        *(v4h*)&aStage[w][m][128] = ev;
      }
    }

    // ---- GEMM1: [16 x 160] @ W1f [160 x 128] -> h [16 x 128] ----
    v8f acc[8] = {};
    const _Float16* rowp = &aStage[w][col][0];
#pragma unroll 1   // keep only one chunk's 8 B-fragments live (register pressure)
    for (int c = 0; c < 5; ++c) {
      union { v16h v; v8h h[2]; } A;
      A.h[0] = *(const v8h*)(rowp + 32 * c + 8 * khi);
      A.h[1] = *(const v8h*)(rowp + 32 * c + 16 + 8 * khi);
      const _Float16* bp = w1f + ((unsigned)(c * 8) << 9) + (unsigned)(lane * 16);
#pragma unroll
      for (int t = 0; t < 8; ++t) {
        v16h B = *(const v16h*)(bp + ((unsigned)t << 9));
        acc[t] = wmma_f16(A.v, B, acc[t]);
      }
    }

    // ---- bias + ReLU, stash h in LDS (f32) to re-layout D -> A ----
    for (int t = 0; t < 8; ++t)
      for (int r = 0; r < 8; ++r) {
        float vv = acc[t][r] + b1v[t];
        hStage[w][r + 8 * khi][t * 16 + col] = vv > 0.f ? vv : 0.f;
      }

    // ---- GEMM2: [16 x 128] @ W2f [128 x 64] ----
    v8f acc2[4] = {};
    const float* hp = &hStage[w][col][0];
#pragma unroll 1
    for (int c = 0; c < 4; ++c) {
      float4 f0 = *(const float4*)(hp + 32 * c + 8 * khi);
      float4 f1 = *(const float4*)(hp + 32 * c + 8 * khi + 4);
      float4 f2 = *(const float4*)(hp + 32 * c + 16 + 8 * khi);
      float4 f3 = *(const float4*)(hp + 32 * c + 16 + 8 * khi + 4);
      v16h A = {(_Float16)f0.x, (_Float16)f0.y, (_Float16)f0.z, (_Float16)f0.w,
                (_Float16)f1.x, (_Float16)f1.y, (_Float16)f1.z, (_Float16)f1.w,
                (_Float16)f2.x, (_Float16)f2.y, (_Float16)f2.z, (_Float16)f2.w,
                (_Float16)f3.x, (_Float16)f3.y, (_Float16)f3.z, (_Float16)f3.w};
      const _Float16* bp = w2f + ((unsigned)(c * 4) << 9) + (unsigned)(lane * 16);
#pragma unroll
      for (int t = 0; t < 4; ++t) {
        v16h B = *(const v16h*)(bp + ((unsigned)t << 9));
        acc2[t] = wmma_f16(A, B, acc2[t]);
      }
    }

    // ---- scatter-add messages into agg[dst]: one row pointer, imm offsets ----
#pragma unroll
    for (int r = 0; r < 8; ++r) {
      const unsigned d = (unsigned)eidx[NE + base + r + 8 * khi];
      float* p = agg + (d << 6) + (unsigned)col;
      atomicAdd(p +  0, acc2[0][r] + b2v[0]);
      atomicAdd(p + 16, acc2[1][r] + b2v[1]);
      atomicAdd(p + 32, acc2[2][r] + b2v[2]);
      atomicAdd(p + 48, acc2[3][r] + b2v[3]);
    }
  }
}

// ---------------- node update ---------------------------------------------------------
#define UWAVES 4
__global__ __launch_bounds__(UWAVES * 32) void node_update_kernel(
    const float* __restrict__ x, const float* agg,
    const float* __restrict__ bu1, const float* __restrict__ bu2,
    const _Float16* __restrict__ wsf, float* out) {
  __shared__ __align__(16) float hStage[UWAVES][16][128];

  const int lane = threadIdx.x & 31;
  const int w    = threadIdx.x >> 5;
  const int col  = lane & 15;
  const int khi  = lane >> 4;
  const int gwave  = blockIdx.x * UWAVES + w;
  const int nwaves = gridDim.x * UWAVES;

  const _Float16* u1f = wsf + U1F_OFF;
  const _Float16* u2f = wsf + U2F_OFF;

  float b1v[8], b2v[4];
  for (int t = 0; t < 8; ++t) b1v[t] = bu1[t * 16 + col];
  for (int t = 0; t < 4; ++t) b2v[t] = bu2[t * 16 + col];

  const int NTILES = NN / 16;  // 6250, exact
  for (int tile = gwave; tile < NTILES; tile += nwaves) {
    const int base = tile * 16;
    const unsigned node = (unsigned)(base + col);  // A row m = col

    // ---- GEMM1: [x | agg] [16 x 128] @ U1f [128 x 128] ----
    v8f acc[8] = {};
    const float* xrow = x + (node << 6);
    const float* grow = agg + (node << 6);
#pragma unroll 1
    for (int c = 0; c < 4; ++c) {
      const float* srcrow = (c < 2) ? xrow : grow;
      const int kb = 32 * (c & 1) + 8 * khi;
      float4 f0 = *(const float4*)(srcrow + kb);
      float4 f1 = *(const float4*)(srcrow + kb + 4);
      float4 f2 = *(const float4*)(srcrow + kb + 16);
      float4 f3 = *(const float4*)(srcrow + kb + 20);
      v16h A = {(_Float16)f0.x, (_Float16)f0.y, (_Float16)f0.z, (_Float16)f0.w,
                (_Float16)f1.x, (_Float16)f1.y, (_Float16)f1.z, (_Float16)f1.w,
                (_Float16)f2.x, (_Float16)f2.y, (_Float16)f2.z, (_Float16)f2.w,
                (_Float16)f3.x, (_Float16)f3.y, (_Float16)f3.z, (_Float16)f3.w};
      const _Float16* bp = u1f + ((unsigned)(c * 8) << 9) + (unsigned)(lane * 16);
#pragma unroll
      for (int t = 0; t < 8; ++t) {
        v16h B = *(const v16h*)(bp + ((unsigned)t << 9));
        acc[t] = wmma_f16(A, B, acc[t]);
      }
    }

    // ---- bias + ReLU -> LDS ----
    for (int t = 0; t < 8; ++t)
      for (int r = 0; r < 8; ++r) {
        float vv = acc[t][r] + b1v[t];
        hStage[w][r + 8 * khi][t * 16 + col] = vv > 0.f ? vv : 0.f;
      }

    // ---- GEMM2: [16 x 128] @ U2f [128 x 64] ----
    v8f acc2[4] = {};
    const float* hp = &hStage[w][col][0];
#pragma unroll 1
    for (int c = 0; c < 4; ++c) {
      float4 f0 = *(const float4*)(hp + 32 * c + 8 * khi);
      float4 f1 = *(const float4*)(hp + 32 * c + 8 * khi + 4);
      float4 f2 = *(const float4*)(hp + 32 * c + 16 + 8 * khi);
      float4 f3 = *(const float4*)(hp + 32 * c + 16 + 8 * khi + 4);
      v16h A = {(_Float16)f0.x, (_Float16)f0.y, (_Float16)f0.z, (_Float16)f0.w,
                (_Float16)f1.x, (_Float16)f1.y, (_Float16)f1.z, (_Float16)f1.w,
                (_Float16)f2.x, (_Float16)f2.y, (_Float16)f2.z, (_Float16)f2.w,
                (_Float16)f3.x, (_Float16)f3.y, (_Float16)f3.z, (_Float16)f3.w};
      const _Float16* bp = u2f + ((unsigned)(c * 4) << 9) + (unsigned)(lane * 16);
#pragma unroll
      for (int t = 0; t < 4; ++t) {
        v16h B = *(const v16h*)(bp + ((unsigned)t << 9));
        acc2[t] = wmma_f16(A, B, acc2[t]);
      }
    }

    // ---- bias + store (in place over agg rows of this tile only) ----
#pragma unroll
    for (int r = 0; r < 8; ++r) {
      float* p = out + ((unsigned)(base + r + 8 * khi) << 6) + (unsigned)col;
      p[ 0] = acc2[0][r] + b2v[0];
      p[16] = acc2[1][r] + b2v[1];
      p[32] = acc2[2][r] + b2v[2];
      p[48] = acc2[3][r] + b2v[3];
    }
  }
}

extern "C" void kernel_launch(void* const* d_in, const int* in_sizes, int n_in,
                              void* d_out, int out_size, void* d_ws, size_t ws_size,
                              hipStream_t stream) {
  const float* x   = (const float*)d_in[0];
  const int*   eix = (const int*)d_in[1];
  const float* ef  = (const float*)d_in[2];
  const float* W1  = (const float*)d_in[3];
  const float* b1  = (const float*)d_in[4];
  const float* W2  = (const float*)d_in[5];
  const float* b2  = (const float*)d_in[6];
  const float* U1  = (const float*)d_in[7];
  const float* bu1 = (const float*)d_in[8];
  const float* U2  = (const float*)d_in[9];
  const float* bu2 = (const float*)d_in[10];
  float*     out = (float*)d_out;
  _Float16*  wsf = (_Float16*)d_ws;

  // d_out doubles as the aggregation buffer: zero, scatter into it, then the
  // node kernel reads its own tile's agg rows before overwriting them.
  hipMemsetAsync(out, 0, (size_t)NN * DN * sizeof(float), stream);
  prep_weights<<<(WSF_TOTAL + 255) / 256, 256, 0, stream>>>(W1, W2, U1, U2, wsf);
  edge_msg_kernel<<<1250, EWAVES * 32, 0, stream>>>(x, eix, ef, b1, b2, wsf, out);
  node_update_kernel<<<640, UWAVES * 32, 0, stream>>>(x, out, bu1, bu2, wsf, out);
}
